// FloatingRegionScore_64510408785941
// MI455X (gfx1250) — matrix operand order repaired and checked by reading, not compiled, not run on hardware
//
#include <hip/hip_runtime.h>
#include <math.h>

// Problem constants (match reference)
#define CC   19
#define HH   1024
#define WW   2048
#define HWN  (HH * WW)
#define RAD  16          // 33x33 window radius
#define PADL 16
#define WP   (WW + 48)   // 16 left pad + 32 right pad
#define LOG19F 2.9444389791664403f
#define EPSF   1e-6f
#define LN2F   0.6931471805599453f
#define LOG2EF 1.4426950408889634f
#define SEG  64          // rows per thread in vertical pass
#define BIAS33 33792.0f  // 1024 * 33 : constant output bias of the magic-f16 trick

typedef __attribute__((ext_vector_type(16))) _Float16 v16h;
typedef __attribute__((ext_vector_type(8)))  _Float16 v8h;
typedef __attribute__((ext_vector_type(8)))  float    v8f;
typedef __attribute__((ext_vector_type(8)))  unsigned int v8u;

// Raw-rate transcendentals: arguments here are always >= 1e-6 (log) and exp2
// underflow to 0 is the desired behavior, so skip libm's denorm pre-scaling.
__device__ __forceinline__ float fast_ln(float x)   { return LN2F * __builtin_amdgcn_logf(x); }
__device__ __forceinline__ float fast_exp(float x)  { return __builtin_amdgcn_exp2f(x * LOG2EF); }

// ---------------- Stage 1: entropy + argmax (pointwise) ----------------
__global__ void k_pointwise(const float* __restrict__ logit,
                            float* __restrict__ ent,
                            unsigned char* __restrict__ pred) {
    int i = blockIdx.x * blockDim.x + threadIdx.x;
    if (i >= HWN) return;
    float v[CC];
    float m = -3.4e38f; int am = 0;
#pragma unroll
    for (int c = 0; c < CC; ++c) {
        float x = logit[(size_t)c * HWN + i];
        v[c] = x;
        if (x > m) { m = x; am = c; }   // first-max like jnp.argmax
    }
    float z = 0.f;
#pragma unroll
    for (int c = 0; c < CC; ++c) { v[c] = fast_exp(v[c] - m); z += v[c]; }
    float invz = 1.0f / z;
    float e = 0.f;
#pragma unroll
    for (int c = 0; c < CC; ++c) {
        float p = v[c] * invz;
        e -= p * fast_ln(p + EPSF);     // p + eps >= 1e-6: denorm path dead
    }
    ent[i]  = e / LOG19F;
    pred[i] = (unsigned char)am;
}

// ---------------- zero the x-padding columns of the padded planes ----------------
__global__ void k_zeropad(_Float16* __restrict__ entVp,
                          unsigned char* __restrict__ cntVp) {
    int i = blockIdx.x * blockDim.x + threadIdx.x;
    const int NPAD = 48;
    int total = 20 * HH * NPAD;
    if (i >= total) return;
    int col  = i % NPAD;
    int rest = i / NPAD;
    int y  = rest % HH;
    int pl = rest / HH;
    int x = (col < PADL) ? col : (WW + col);  // cols [0,16) and [W+16, W+48)
    if (pl == CC) entVp[(size_t)y * WP + x] = (_Float16)0.0f;
    else          cntVp[(size_t)pl * (HH * (size_t)WP) + (size_t)y * WP + x] = 0;
}

// ---------------- Stage 2: vertical box sums (sliding window per column) ----------------
__global__ void k_vbox(const float* __restrict__ ent,
                       const unsigned char* __restrict__ pred,
                       _Float16* __restrict__ entVp,
                       unsigned char* __restrict__ cntVp) {
    int tid = blockIdx.x * blockDim.x + threadIdx.x;
    const int nseg = HH / SEG;
    if (tid >= WW * nseg) return;
    int x  = tid % WW;
    int y0 = (tid / WW) * SEG;

    int cnt[CC];
#pragma unroll
    for (int c = 0; c < CC; ++c) cnt[c] = 0;
    float es = 0.f;

    for (int r = y0 - RAD; r <= y0 + RAD; ++r) {
        if (r < 0 || r >= HH) continue;
        es += ent[(size_t)r * WW + x];
        int pv = (int)pred[(size_t)r * WW + x];
#pragma unroll
        for (int c = 0; c < CC; ++c) cnt[c] += (pv == c);
    }
    for (int y = y0; y < y0 + SEG; ++y) {
        size_t base = (size_t)y * WP + PADL + x;
        entVp[base] = (_Float16)es;
#pragma unroll
        for (int c = 0; c < CC; ++c)
            cntVp[(size_t)c * (HH * (size_t)WP) + base] = (unsigned char)cnt[c];
        int rin = y + RAD + 1, rout = y - RAD;
        if (rin < HH) {
            es += ent[(size_t)rin * WW + x];
            int pv = (int)pred[(size_t)rin * WW + x];
#pragma unroll
            for (int c = 0; c < CC; ++c) cnt[c] += (pv == c);
        }
        if (rout >= 0) {
            es -= ent[(size_t)rout * WW + x];
            int pv = (int)pred[(size_t)rout * WW + x];
#pragma unroll
            for (int c = 0; c < CC; ++c) cnt[c] -= (pv == c);
        }
    }
}

// ---------------- Stage 3 helpers: build WMMA A operands ----------------
// A (16-bit, 16x32): lane m = L&15; K base = (L>=16)?8:0; elems 0..7 -> K=kb+0..7,
// elems 8..15 -> K=kb+16..23  (CDNA5 ISA, 16-bit A 16x32 table)

// Magic-bias u8 -> f16: half(1024+n) == 0x6400|n for n<1024 (exact).
// Built with v_perm_b32: result bytes [b, 0x64, b', 0x64] from (src0=0x64646464, src1=w).
__device__ __forceinline__ v16h load_a_u8_biased(const unsigned char* plane, int row,
                                                 int col0, int kb) {
    const unsigned char* p = plane + (size_t)row * WP + col0 + kb;
    uint2 lo = *(const uint2*)(p);
    uint2 hh = *(const uint2*)(p + 16);
    unsigned int w[4] = { lo.x, lo.y, hh.x, hh.y };
    const unsigned int K64 = 0x64646464u;
    v8u r;
#pragma unroll
    for (int j = 0; j < 4; ++j) {
        // bytes (b0,b1) -> halves 2j ; bytes (b2,b3) -> halves 2j+1
        r[2 * j]     = __builtin_amdgcn_perm(K64, w[j], 0x04010400u);
        r[2 * j + 1] = __builtin_amdgcn_perm(K64, w[j], 0x04030402u);
    }
    union { v8u u; v16h h; } cv; cv.u = r;
    return cv.h;
}

__device__ __forceinline__ v16h load_a_f16(const _Float16* plane, int row, int col0, int kb) {
    const _Float16* p = plane + (size_t)row * WP + col0 + kb;
    v8h lo = *(const v8h*)(p);
    v8h hh = *(const v8h*)(p + 16);
    v16h a;
#pragma unroll
    for (int e = 0; e < 8; ++e) { a[e] = lo[e]; a[e + 8] = hh[e]; }
    return a;
}

// ---------------- Stage 3: horizontal box via WMMA + final score ----------------
// Out(16y x 16x) = A(16y x 48k) * S(48k x 16x); split K as 32 + 16(zero-padded).
// S1[K][n] = (K >= n), K in [0,31] (32-n ones); S2[K][n] = (K < 16 && K <= n) (n+1 ones).
// Biased count inputs add exactly 1024*33 = 33792 to every output element.
__global__ void k_hbox(const _Float16* __restrict__ entVp,
                       const unsigned char* __restrict__ cntVp,
                       float* __restrict__ out) {
    int wave = (blockIdx.x * blockDim.x + threadIdx.x) >> 5;
    int lane = threadIdx.x & 31;
    const int TX = WW / 16, TY = HH / 16;
    if (wave >= TX * TY) return;               // grid sized exactly; no divergence
    int tx = wave % TX, ty = wave / TX;
    int x0 = tx * 16, y0 = ty * 16;

    int n   = lane & 15;
    int hi  = lane >> 4;
    int kb  = hi ? 8 : 0;    // A-operand K base
    int kbB = hi ? 16 : 0;   // unused: B (s1, s2) is built with kb, same k order as A

    // Banded selection matrices, constant per lane (built in registers)
    v16h s1, s2;
#pragma unroll
    for (int e = 0; e < 16; ++e) {
        int K = kb + e + ((e >= 8) ? 8 : 0);
        s1[e] = (_Float16)((K >= n) ? 1.0f : 0.0f);
        s2[e] = (_Float16)((K < 16 && K <= n) ? 1.0f : 0.0f);
    }

    int arow = y0 + (lane & 15);  // A-operand row for this lane

    // analytic count = wh * ww (box filter of ones with zero padding)
    int x  = x0 + n;
    int ww = min(x + RAD, WW - 1) - max(x - RAD, 0) + 1;
    float invcnt[8];
#pragma unroll
    for (int v = 0; v < 8; ++v) {
        int y  = y0 + v + (hi ? 8 : 0);        // D layout: VGPR v -> M = v + 8*hi
        int wh = min(y + RAD, HH - 1) - max(y - RAD, 0) + 1;
        invcnt[v] = 1.0f / (float)(wh * ww);
    }

    float imp[8];
#pragma unroll
    for (int v = 0; v < 8; ++v) imp[v] = 0.f;

    // 19 class-count channels (u8 planes, magic-biased into f16; exact arithmetic)
    for (int c = 0; c < CC; ++c) {
        const unsigned char* pl = cntVp + (size_t)c * (HH * (size_t)WP);
        v16h a1 = load_a_u8_biased(pl, arow, x0, kb);
        v16h a2 = load_a_u8_biased(pl, arow, x0 + 32, kb);
        v8f acc = {};
        acc = __builtin_amdgcn_wmma_f32_16x16x32_f16(false, a1, false, s1, (short)0, acc, false, false);
        acc = __builtin_amdgcn_wmma_f32_16x16x32_f16(false, a2, false, s2, (short)0, acc, false, false);
#pragma unroll
        for (int v = 0; v < 8; ++v) {
            float s = acc[v] - BIAS33;         // remove constant magic bias
            float d = s * invcnt[v];
            imp[v] -= d * fast_ln(d + EPSF);   // d + eps >= 1e-6: denorm path dead
        }
    }

    // entropy channel (f16 plane, unbiased)
    v16h e1 = load_a_f16(entVp, arow, x0, kb);
    v16h e2 = load_a_f16(entVp, arow, x0 + 32, kb);
    v8f ea = {};
    ea = __builtin_amdgcn_wmma_f32_16x16x32_f16(false, e1, false, s1, (short)0, ea, false, false);
    ea = __builtin_amdgcn_wmma_f32_16x16x32_f16(false, e2, false, s2, (short)0, ea, false, false);

#pragma unroll
    for (int v = 0; v < 8; ++v) {
        int y = y0 + v + (hi ? 8 : 0);
        float ri = imp[v] / LOG19F;
        float pu = ea[v] * invcnt[v];
        size_t o = (size_t)y * WW + x;
        out[o]               = ri * pu;  // score
        out[(size_t)HWN + o] = ri;       // region_impurity
        out[2*(size_t)HWN+o] = pu;       // prediction_uncertainty
    }
}

// ---------------- launch ----------------
extern "C" void kernel_launch(void* const* d_in, const int* in_sizes, int n_in,
                              void* d_out, int out_size, void* d_ws, size_t ws_size,
                              hipStream_t stream) {
    const float* logit = (const float*)d_in[0];
    float* out = (float*)d_out;
    char* ws = (char*)d_ws;

    // L2-resident scratch layout (all 256B aligned), total ~53 MB
    float*         ent   = (float*)(ws + 0);                       //  8,388,608 B
    unsigned char* pred  = (unsigned char*)(ws + 8388608);         //  2,097,152 B
    _Float16*      entVp = (_Float16*)(ws + 10485760);             //  4,292,608 B
    unsigned char* cntVp = (unsigned char*)(ws + 14778368);        // 40,779,776 B

    // Stage 1: pointwise softmax entropy + argmax
    k_pointwise<<<(HWN + 255) / 256, 256, 0, stream>>>(logit, ent, pred);

    // Zero the horizontal padding columns (re-done every call: ws is poisoned once)
    {
        int total = 20 * HH * 48;
        k_zeropad<<<(total + 255) / 256, 256, 0, stream>>>(entVp, cntVp);
    }

    // Stage 2: vertical 33-row sliding box sums
    {
        int threads = WW * (HH / SEG);
        k_vbox<<<(threads + 255) / 256, 256, 0, stream>>>(ent, pred, entVp, cntVp);
    }

    // Stage 3: horizontal 33-col box via WMMA + final score (one wave per 16x16 tile)
    {
        int waves = (WW / 16) * (HH / 16);      // 8192
        k_hbox<<<(waves * 32) / 256, 256, 0, stream>>>(entVp, cntVp, out);
    }
}
